// Attention_42958262895140
// MI455X (gfx1250) — compile-verified, matrix operand order checked by reading, not run on hardware
//
#include <hip/hip_runtime.h>
#include <hip/hip_bf16.h>

#define HEAD_NUM 16
#define HEAD_DIM 128
#define IN_DIM   1024
#define OUT_DIM  1024
#define QKV_N    (3 * HEAD_NUM * HEAD_DIM)   // 6144
#define TOK_TILE 16
#define THREADS  512
#define NWAVES   (THREADS / 32)

#define XSTR 1032   // padded x row stride (bf16 elems): 2064B -> conflict-free, 16B aligned
#define HSTR 136    // padded per-head row stride: 272B -> conflict-free, 16B aligned

typedef __bf16 bf16_t;
typedef __attribute__((ext_vector_type(16))) __bf16 v16bf;
typedef __attribute__((ext_vector_type(8)))  float  v8f;
typedef __attribute__((ext_vector_type(4)))  unsigned int u32x4;

union Frag { v16bf v; u32x4 q[2]; };

// Load one WMMA 16x32 bf16 operand from a row-major [16 x K] slab.
// Matches the CDNA5 16-bit A/B VGPR layout: lane L -> row L%16; lane-half h=L/16
// supplies K = {k0+8h .. k0+8h+7} in VGPRs 0-3 and K = {k0+16+8h ..} in VGPRs 4-7.
__device__ __forceinline__ v16bf load_frag(const bf16_t* __restrict__ base,
                                           int rstride, int k0, int lane) {
  const int r = lane & 15;
  const int h = lane >> 4;
  const bf16_t* p0 = base + (size_t)r * rstride + k0 + h * 8;
  Frag f;
  f.q[0] = *(const u32x4*)(p0);
  f.q[1] = *(const u32x4*)(p0 + 16);
  return f.v;
}

#define WMMA_BF16(a, b, c) \
  __builtin_amdgcn_wmma_f32_16x16x32_bf16(false, (a), false, (b), (short)0, (c), false, false)

// Software-pipelined dual-column-tile GEMM K-loop: one shared A fragment,
// two independent B streams / accumulator chains.  Unroll-by-2 ping-pong with
// consume-then-refill ordering: each buffer set's WMMAs issue first, then the
// set is refilled for step kb+2/kb+3, giving every load a full iteration of
// load-to-use distance (the waits in iteration i cover loads issued in i-1).
// Requires even nkb >= 2.
__device__ __forceinline__ void gemm_dual(const bf16_t* __restrict__ abase, int astride,
                                          const bf16_t* __restrict__ wb0,
                                          const bf16_t* __restrict__ wb1,
                                          int wstride, int nkb, int lane,
                                          v8f& acc0, v8f& acc1) {
  v16bf a0 = load_frag(abase, astride, 0, lane);
  v16bf p0 = load_frag(wb0,   wstride, 0, lane);
  v16bf q0 = load_frag(wb1,   wstride, 0, lane);
  v16bf a1 = load_frag(abase, astride, 32, lane);
  v16bf p1 = load_frag(wb0,   wstride, 32, lane);
  v16bf q1 = load_frag(wb1,   wstride, 32, lane);
  #pragma unroll 1
  for (int kb = 0; kb + 2 < nkb; kb += 2) {
    acc0 = WMMA_BF16(a0, p0, acc0);
    acc1 = WMMA_BF16(a0, q0, acc1);
    a0 = load_frag(abase, astride, (kb + 2) * 32, lane);
    p0 = load_frag(wb0,   wstride, (kb + 2) * 32, lane);
    q0 = load_frag(wb1,   wstride, (kb + 2) * 32, lane);
    acc0 = WMMA_BF16(a1, p1, acc0);
    acc1 = WMMA_BF16(a1, q1, acc1);
    a1 = load_frag(abase, astride, (kb + 3) * 32, lane);
    p1 = load_frag(wb0,   wstride, (kb + 3) * 32, lane);
    q1 = load_frag(wb1,   wstride, (kb + 3) * 32, lane);
  }
  // epilogue: steps nkb-2 / nkb-1 are already resident in the two sets
  acc0 = WMMA_BF16(a0, p0, acc0);
  acc1 = WMMA_BF16(a0, q0, acc1);
  acc0 = WMMA_BF16(a1, p1, acc0);
  acc1 = WMMA_BF16(a1, q1, acc1);
}

// One-time weight prep: f32 -> bf16 with transpose to column-major so WMMA B
// operands (B[k][n] = W[k][col0+n] = Wt[col0+n][k]) are two contiguous 16B loads.
__global__ void prep_weights(const float* __restrict__ W_pre,
                             const float* __restrict__ W_proj,
                             bf16_t* __restrict__ WqkvT,    // [6144][1024]
                             bf16_t* __restrict__ WprojT) { // [1024][128]
  const size_t n1 = (size_t)IN_DIM * QKV_N;
  const size_t n2 = (size_t)HEAD_DIM * OUT_DIM;
  for (size_t i = (size_t)blockIdx.x * blockDim.x + threadIdx.x; i < n1 + n2;
       i += (size_t)gridDim.x * blockDim.x) {
    if (i < n1) {
      size_t k = i / QKV_N, c = i % QKV_N;          // coalesced read
      WqkvT[c * IN_DIM + k] = (bf16_t)W_pre[i];
    } else {
      size_t j = i - n1;
      size_t d = j / OUT_DIM, o = j % OUT_DIM;      // coalesced read
      WprojT[o * HEAD_DIM + d] = (bf16_t)W_proj[j];
    }
  }
}

__global__ __launch_bounds__(THREADS, 1)
void fused_attn(const float* __restrict__ x,
                const bf16_t* __restrict__ WqkvT,
                const float* __restrict__ b_pre,
                const bf16_t* __restrict__ WprojT,
                const float* __restrict__ b_proj,
                float* __restrict__ out) {
  __shared__ bf16_t sx[TOK_TILE * XSTR];               // x tile, bf16
  __shared__ bf16_t sq[TOK_TILE * HEAD_NUM * HSTR];    // Q tile  [tok][h][d]
  __shared__ bf16_t sk[TOK_TILE * HEAD_NUM * HSTR];    // K tile  [tok][h][d]
  __shared__ float  ssum[TOK_TILE * 16];               // sum_h softmax(score)  [tok][g]
  __shared__ float  satt[TOK_TILE * HEAD_DIM];         // att summed over heads [tok][d]
  __shared__ bf16_t sattb[TOK_TILE * HSTR];            // bf16 copy for phase 4

  const int tid   = threadIdx.x;
  const int lane  = tid & 31;
  const int wave  = tid >> 5;
  const int tok0  = blockIdx.x * TOK_TILE;
  const int colL  = lane & 15;                         // C/D tile: column = lane%16
  const int rbase = (lane >> 4) << 3;                  // C/D tile: rows rbase..rbase+7

  // ---- stage x tile: f32 global -> bf16 LDS (coalesced) ----
  for (int i = tid; i < TOK_TILE * IN_DIM; i += THREADS) {
    int r = i >> 10, c = i & (IN_DIM - 1);
    sx[r * XSTR + c] = (bf16_t)x[(size_t)(tok0 + r) * IN_DIM + c];
  }
  __syncthreads();

  // ---- phase 1: Q,K projection  [16 x 1024] @ [1024 x 4096] ----
  // 128 column-tile pairs; pairs never straddle the Q|K boundary (pair base is
  // a multiple of 32 columns, halves are 2048-aligned).
  for (int p = wave; p < 128; p += NWAVES) {
    const int c0 = p * 32;
    const int c1 = c0 + 16;
    v8f acc0 = {}, acc1 = {};
    gemm_dual(sx, XSTR,
              WqkvT + (size_t)c0 * IN_DIM, WqkvT + (size_t)c1 * IN_DIM,
              IN_DIM, IN_DIM / 32, lane, acc0, acc1);
    bf16_t* dst = (c0 < 2048) ? sq : sk;
    const int cbase = c0 & 2047;
    #pragma unroll
    for (int half = 0; half < 2; ++half) {
      const int cc = cbase + half * 16 + colL;
      const int h = cc >> 7, d = cc & 127;
      const float bias = b_pre[(c0 + half * 16) + colL];
      const v8f& acc = half ? acc1 : acc0;
      #pragma unroll
      for (int r = 0; r < 8; ++r)
        dst[(size_t)(r + rbase) * (HEAD_NUM * HSTR) + h * HSTR + d] = (bf16_t)(acc[r] + bias);
    }
  }
  __syncthreads();

  // ---- phase 2: per-token scores (16h x 16g over D=128), softmax, head-sum ----
  {
    const int b = wave;                                // one wave per token
    const bf16_t* qb  = sq + (size_t)b * HEAD_NUM * HSTR;
    const bf16_t* kbp = sk + (size_t)b * HEAD_NUM * HSTR;
    v8f acc = {};
    #pragma unroll
    for (int kb = 0; kb < HEAD_DIM / 32; ++kb) {
      v16bf a  = load_frag(qb,  HSTR, kb * 32, lane);  // A: rows = heads h
      v16bf bb = load_frag(kbp, HSTR, kb * 32, lane);  // B: cols = heads g (k^T)
      acc = WMMA_BF16(a, bb, acc);
    }
    const float scale = 0.08838834764831845f;          // 1/sqrt(128)
    float csum = 0.f;
    #pragma unroll
    for (int r = 0; r < 8; ++r) {
      float s = acc[r] * scale;
      float m = s;                                     // row max across the 16-lane group
      m = fmaxf(m, __shfl_xor(m, 1));
      m = fmaxf(m, __shfl_xor(m, 2));
      m = fmaxf(m, __shfl_xor(m, 4));
      m = fmaxf(m, __shfl_xor(m, 8));
      float e  = __expf(s - m);
      float sm = e;                                    // row sum
      sm += __shfl_xor(sm, 1);
      sm += __shfl_xor(sm, 2);
      sm += __shfl_xor(sm, 4);
      sm += __shfl_xor(sm, 8);
      csum += e / sm;                                  // accumulate column sum over rows
    }
    csum += __shfl_xor(csum, 16);                      // fold rows 8..15 into 0..7
    if (lane < 16) ssum[b * 16 + lane] = csum;
  }
  __syncthreads();

  // ---- phase 3: V projection fused with score-weighted head reduction ----
  // wave -> (d-chunk, head-half); 4 head-pairs per wave, register accumulation
  // keeps the f32 sum order fixed (deterministic, no float atomics).
  {
    const int dc = wave & 7;
    const int gh = wave >> 3;
    float att[8] = {0.f, 0.f, 0.f, 0.f, 0.f, 0.f, 0.f, 0.f};
    #pragma unroll 1
    for (int gi = 0; gi < 4; ++gi) {
      const int g0 = gh * 8 + gi * 2;
      const int g1 = g0 + 1;
      const int c0 = 4096 + g0 * HEAD_DIM + dc * 16;   // V columns in W_pre
      const int c1 = 4096 + g1 * HEAD_DIM + dc * 16;
      v8f acc0 = {}, acc1 = {};
      gemm_dual(sx, XSTR,
                WqkvT + (size_t)c0 * IN_DIM, WqkvT + (size_t)c1 * IN_DIM,
                IN_DIM, IN_DIM / 32, lane, acc0, acc1);
      const float bias0 = b_pre[c0 + colL];
      const float bias1 = b_pre[c1 + colL];
      #pragma unroll
      for (int r = 0; r < 8; ++r) {
        const float* srow = &ssum[(r + rbase) * 16];
        att[r] += srow[g0] * (acc0[r] + bias0) + srow[g1] * (acc1[r] + bias1);
      }
    }
    if (gh == 0) {
      #pragma unroll
      for (int r = 0; r < 8; ++r)
        satt[(r + rbase) * HEAD_DIM + dc * 16 + colL] = att[r];
    }
    __syncthreads();
    if (gh == 1) {
      #pragma unroll
      for (int r = 0; r < 8; ++r)
        satt[(r + rbase) * HEAD_DIM + dc * 16 + colL] += att[r];
    }
  }
  __syncthreads();

  // ---- phase 4: output projection  [16 x 128] @ [128 x 1024] + 16*b_proj ----
  for (int i = tid; i < TOK_TILE * HEAD_DIM; i += THREADS) {
    int r = i >> 7, d = i & 127;
    sattb[r * HSTR + d] = (bf16_t)satt[i];
  }
  __syncthreads();
  for (int p = wave; p < OUT_DIM / 32; p += NWAVES) {
    const int c0 = p * 32;
    const int c1 = c0 + 16;
    v8f acc0 = {}, acc1 = {};
    gemm_dual(sattb, HSTR,
              WprojT + (size_t)c0 * HEAD_DIM, WprojT + (size_t)c1 * HEAD_DIM,
              HEAD_DIM, HEAD_DIM / 32, lane, acc0, acc1);
    #pragma unroll
    for (int half = 0; half < 2; ++half) {
      const int c = c0 + half * 16 + colL;
      const float bias = 16.f * b_proj[c];             // bias added per head, then summed
      const v8f& acc = half ? acc1 : acc0;
      #pragma unroll
      for (int r = 0; r < 8; ++r)
        out[(size_t)(tok0 + r + rbase) * OUT_DIM + c] = acc[r] + bias;
    }
  }
}

extern "C" void kernel_launch(void* const* d_in, const int* in_sizes, int n_in,
                              void* d_out, int out_size, void* d_ws, size_t ws_size,
                              hipStream_t stream) {
  const float* x      = (const float*)d_in[0];
  const float* W_pre  = (const float*)d_in[1];
  const float* b_pre  = (const float*)d_in[2];
  const float* W_proj = (const float*)d_in[3];
  const float* b_proj = (const float*)d_in[4];
  float* out = (float*)d_out;

  // workspace: bf16 W_pre^T (12.6 MB) + bf16 W_proj^T (0.25 MB)
  bf16_t* WqkvT  = (bf16_t*)d_ws;
  bf16_t* WprojT = WqkvT + (size_t)IN_DIM * QKV_N;

  prep_weights<<<512, 256, 0, stream>>>(W_pre, W_proj, WqkvT, WprojT);

  const int B = 64 * 512;
  fused_attn<<<B / TOK_TILE, THREADS, 0, stream>>>(x, WqkvT, b_pre, WprojT, b_proj, out);
}
